// ElementalModesMessagePassingNeuralNetwork_78924319031914
// MI455X (gfx1250) — matrix operand order repaired and checked by reading, not compiled, not run on hardware
//
#include <hip/hip_runtime.h>
#include <hip/hip_bf16.h>

// ---------------------------------------------------------------------------
// Types for CDNA5 WMMA (wave32, 16x16x32 bf16 -> f32)
// ---------------------------------------------------------------------------
typedef __bf16 v16bf __attribute__((ext_vector_type(16)));
typedef float  v8f   __attribute__((ext_vector_type(8)));

union ABfrag { v16bf v; uint4 u[2]; };

__device__ __forceinline__ unsigned short f2bf(float f) {
    unsigned u = __float_as_uint(f);
    u += 0x7FFFu + ((u >> 16) & 1u);          // round-to-nearest-even
    return (unsigned short)(u >> 16);
}

// shifted softplus: log(1+exp(x)) - log(2), numerically stable
__device__ __forceinline__ float sspf(float x) {
    float ax = fabsf(x);
    return fmaxf(x, 0.f) + log1pf(expf(-ax)) - 0.6931471805599453f;
}

enum { F_SSP_IN = 1, F_SSP_OUT = 2, F_RES = 4, F_RES_IU = 8 };

#define LDT 136   // padded LDS row stride (bf16 elements) to spread banks

// ---------------------------------------------------------------------------
// WMMA GEMM: Y[M,N] = post( pre(X[M,K]) @ W[K,N(ldw)] + bias ) (+res / +iu*res)
// Tile: 128x128 per workgroup, 8 waves, each wave a 16-row strip.
// ---------------------------------------------------------------------------
__global__ __launch_bounds__(256)
void gemm_wmma_kernel(const float* __restrict__ X, const float* __restrict__ W,
                      const float* __restrict__ bias, const float* __restrict__ res,
                      const float* __restrict__ iuv, float* __restrict__ Y,
                      int M, int N, int K, int ldw, int ldy, int flags)
{
    __shared__ unsigned short sA[128 * LDT];
    __shared__ unsigned short sB[128 * LDT];

    const int t = threadIdx.x;
    const int rowBase = blockIdx.x * 128;
    const int colBase = blockIdx.y * 128;

    // ---- stage A (rows of X, optional ssp, f32 -> bf16) ----
    const int kv4 = K >> 2;
    for (int idx = t; idx < 128 * kv4; idx += 256) {
        int r  = idx / kv4;
        int k4 = idx - r * kv4;
        int rg = rowBase + r;
        float4 v = make_float4(0.f, 0.f, 0.f, 0.f);
        if (rg < M) v = *(const float4*)(X + (size_t)rg * K + (size_t)k4 * 4);
        if (flags & F_SSP_IN) { v.x = sspf(v.x); v.y = sspf(v.y); v.z = sspf(v.z); v.w = sspf(v.w); }
        unsigned short* p = &sA[r * LDT + k4 * 4];
        p[0] = f2bf(v.x); p[1] = f2bf(v.y); p[2] = f2bf(v.z); p[3] = f2bf(v.w);
    }
    // ---- stage B transposed: sB[n][k] = bf16(W[k][colBase+n]), zero-pad ----
    for (int idx = t; idx < 128 * K; idx += 256) {
        int k = idx >> 7;
        int n = idx & 127;
        int cg = colBase + n;
        float v = (cg < N) ? W[(size_t)k * ldw + cg] : 0.f;
        sB[n * LDT + k] = f2bf(v);
    }
    __syncthreads();

    const int w    = t >> 5;
    const int lane = t & 31;
    const int hi   = lane >> 4;   // 0 or 1 (lane half)
    const int l15  = lane & 15;

    v8f acc[8];
#pragma unroll
    for (int nt = 0; nt < 8; ++nt)
#pragma unroll
        for (int q = 0; q < 8; ++q) acc[nt][q] = 0.f;

    const int arow = (w << 4) + l15;          // A row within tile (M = lane&15)
    const int kch  = K >> 5;                  // K chunks of 32

    for (int kc = 0; kc < kch; ++kc) {
        ABfrag a;
        // A 16x32 bf16 layout: lanes<16 -> K {0..7,16..23}; lanes>=16 -> {8..15,24..31}
        int ka = kc * 32 + (hi << 3);
        a.u[0] = *(const uint4*)&sA[arow * LDT + ka];
        a.u[1] = *(const uint4*)&sA[arow * LDT + ka + 16];
#pragma unroll
        for (int nt = 0; nt < 8; ++nt) {
            ABfrag b;
            // B 32x16 bf16 layout: lanes<16 -> K {0..7,16..23}; lanes>=16 -> {8..15,24..31}; N = lane&15
            int col = (nt << 4) + l15;
            int kb  = kc * 32 + (hi << 3);
            b.u[0] = *(const uint4*)&sB[col * LDT + kb];
            b.u[1] = *(const uint4*)&sB[col * LDT + kb + 16];
            acc[nt] = __builtin_amdgcn_wmma_f32_16x16x32_bf16(
                false, a.v, false, b.v, (short)0, acc[nt], false, false);
        }
    }

    // ---- epilogue: C 16x16 f32 layout: VGPR i -> M = i + 8*hi, N = lane&15 ----
#pragma unroll
    for (int nt = 0; nt < 8; ++nt) {
        int colg = colBase + (nt << 4) + l15;
        if (colg >= N) continue;
        float bv = bias ? bias[colg] : 0.f;
#pragma unroll
        for (int i = 0; i < 8; ++i) {
            int rowg = rowBase + (w << 4) + (hi << 3) + i;
            if (rowg >= M) continue;
            float v = acc[nt][i] + bv;
            if (flags & F_SSP_OUT) v = sspf(v);
            if (flags & F_RES) {
                float rv = res[(size_t)rowg * 128 + colg];
                if (flags & F_RES_IU) rv *= iuv[colg];
                v += rv;
            }
            Y[(size_t)rowg * ldy + colg] = v;
        }
    }
}

// ---------------------------------------------------------------------------
// Edge kernel: Dij + cutoff + 64 RBF features per edge
// ---------------------------------------------------------------------------
__global__ __launch_bounds__(256)
void edge_rbf_kernel(const float* __restrict__ R, const int* __restrict__ idx_i,
                     const int* __restrict__ idx_j, float* __restrict__ rbf,
                     float* __restrict__ dij_out, int E)
{
    int e = blockIdx.x * 256 + threadIdx.x;
    if (e >= E) return;
    int i = idx_i[e], j = idx_j[e];
    float dx = R[i * 3 + 0] - R[j * 3 + 0];
    float dy = R[i * 3 + 1] - R[j * 3 + 1];
    float dz = R[i * 3 + 2] - R[j * 3 + 2];
    float d2 = dx * dx + dy * dy + dz * dz;
    float dij = sqrtf(fmaxf(d2, 0.f));
    dij_out[e] = dij;

    float tt = dij * 0.1f;
    float t2 = tt * tt, t3 = t2 * tt, t4 = t3 * tt, t5 = t4 * tt;
    float cut = (dij < 10.f) ? (1.f - 6.f * t5 + 15.f * t4 - 10.f * t3) : 0.f;
    float de = expf(-dij);

    const float mu0 = 4.5399929762484854e-05f;        // exp(-CUTOFF)
    const float dmu = (1.f - mu0) / 63.f;
    const float gt  = 64.f / (2.f * (1.f - mu0));
    const float gamma = gt * gt;

    float* rp = rbf + (size_t)e * 64;
#pragma unroll 4
    for (int k = 0; k < 64; ++k) {
        float mu = mu0 + (float)k * dmu;
        float dd = de - mu;
        rp[k] = cut * expf(-gamma * dd * dd);
    }
}

// ---------------------------------------------------------------------------
// Embedding first layer: h = ssp([Z,M,QaA,QaB] @ W0 + b0), K=4
// ---------------------------------------------------------------------------
__global__ __launch_bounds__(256)
void embed_kernel(const int* __restrict__ Z, const float* __restrict__ Mm,
                  const float* __restrict__ QaA, const float* __restrict__ QaB,
                  const float* __restrict__ W0, const float* __restrict__ b0,
                  float* __restrict__ h, int N)
{
    int idx = blockIdx.x * 256 + threadIdx.x;
    if (idx >= N * 128) return;
    int a = idx >> 7, f = idx & 127;
    float s = (float)Z[a] * W0[f] + Mm[a] * W0[128 + f]
            + QaA[a] * W0[256 + f] + QaB[a] * W0[384 + f] + b0[f];
    h[idx] = sspf(s);
}

// ---------------------------------------------------------------------------
// Message: m[a] += sum_{e in atom a's 32 edges} g[e] * xj[idx_j[e]]
// idx_i is sorted -> contiguous edge ranges, no atomics needed.
// ---------------------------------------------------------------------------
__global__ __launch_bounds__(128)
void message_kernel(float* __restrict__ m, const float* __restrict__ g,
                    const float* __restrict__ xj, const int* __restrict__ idx_j, int N)
{
    __shared__ int sj[32];
    int a = blockIdx.x;
    int f = threadIdx.x;
    if (f < 32) sj[f] = idx_j[(size_t)a * 32 + f];
    __syncthreads();
    float acc = m[(size_t)a * 128 + f];
    const float* gp = g + (size_t)a * 32 * 128 + f;
#pragma unroll 4
    for (int e = 0; e < 32; ++e)
        acc += gp[(size_t)e * 128] * xj[(size_t)sj[e] * 128 + f];
    m[(size_t)a * 128 + f] = acc;
}

// ---------------------------------------------------------------------------
// Per-molecule segment sums (mol_idx = (a*NMOL)/N, sorted -> closed-form range)
// ---------------------------------------------------------------------------
__global__ __launch_bounds__(256)
void segsum2_kernel(const float* __restrict__ A, const float* __restrict__ B,
                    float* __restrict__ out, int N, int NMOLc)
{
    int mres = blockIdx.x;
    int start = (int)(((long long)mres * N + NMOLc - 1) / NMOLc);
    int end   = (int)(((long long)(mres + 1) * N + NMOLc - 1) / NMOLc);
    float sa = 0.f, sb = 0.f;
    for (int a = start + threadIdx.x; a < end; a += 256) { sa += A[a]; sb += B[a]; }
    __shared__ float ra[256], rb[256];
    ra[threadIdx.x] = sa; rb[threadIdx.x] = sb; __syncthreads();
    for (int s = 128; s > 0; s >>= 1) {
        if (threadIdx.x < (unsigned)s) { ra[threadIdx.x] += ra[threadIdx.x + s]; rb[threadIdx.x] += rb[threadIdx.x + s]; }
        __syncthreads();
    }
    if (threadIdx.x == 0) { out[mres] = ra[0]; out[NMOLc + mres] = rb[0]; }
}

__global__ __launch_bounds__(256)
void segsum_cols_kernel(const float* __restrict__ O, int ldo,
                        float* __restrict__ out4, int N, int NMOLc)
{
    int mres = blockIdx.x;
    int start = (int)(((long long)mres * N + NMOLc - 1) / NMOLc);
    int end   = (int)(((long long)(mres + 1) * N + NMOLc - 1) / NMOLc);
    float s0 = 0.f, s1 = 0.f, s2 = 0.f, s3 = 0.f;
    for (int a = start + threadIdx.x; a < end; a += 256) {
        const float* p = O + (size_t)a * ldo;
        s0 += p[0]; s1 += p[1]; s2 += p[2]; s3 += p[3];
    }
    __shared__ float r0[256], r1[256], r2[256], r3[256];
    r0[threadIdx.x] = s0; r1[threadIdx.x] = s1; r2[threadIdx.x] = s2; r3[threadIdx.x] = s3;
    __syncthreads();
    for (int s = 128; s > 0; s >>= 1) {
        if (threadIdx.x < (unsigned)s) {
            r0[threadIdx.x] += r0[threadIdx.x + s]; r1[threadIdx.x] += r1[threadIdx.x + s];
            r2[threadIdx.x] += r2[threadIdx.x + s]; r3[threadIdx.x] += r3[threadIdx.x + s];
        }
        __syncthreads();
    }
    if (threadIdx.x == 0) {
        out4[mres * 4 + 0] = r0[0]; out4[mres * 4 + 1] = r1[0];
        out4[mres * 4 + 2] = r2[0]; out4[mres * 4 + 3] = r3[0];
    }
}

// SCC charge re-balancing
__global__ __launch_bounds__(256)
void scc_update_kernel(float* __restrict__ QaA, float* __restrict__ QaB,
                       const float* __restrict__ O, int ldo,
                       const float* __restrict__ molAB, const float* __restrict__ molT,
                       const int* __restrict__ mol_idx, int N, int NMOLc)
{
    int a = blockIdx.x * 256 + threadIdx.x;
    if (a >= N) return;
    int m = mol_idx[a];
    const float* p = O + (size_t)a * ldo;
    float qa = p[0], fa = p[1], qb = p[2], fb = p[3];
    QaA[a] = qa + fa * (molAB[m] - molT[m * 4 + 0]) / molT[m * 4 + 1];
    QaB[a] = qb + fb * (molAB[NMOLc + m] - molT[m * 4 + 2]) / molT[m * 4 + 3];
}

// outputs += out; nh-loss partials (fixed-order, deterministic)
__global__ __launch_bounds__(256)
void accum_out_kernel(float* __restrict__ outs, const float* __restrict__ tmp,
                      float* __restrict__ prev2, float* __restrict__ partials,
                      int total, int ldo, int doNH, int useNH)
{
    int idx = blockIdx.x * 256 + threadIdx.x;
    float r = 0.f;
    if (idx < total) {
        float v = tmp[idx];
        outs[idx] += v;
        if (doNH) {
            int a = idx / ldo;
            if (a >= 4) {
                float o2 = v * v;
                if (useNH) r = o2 / (o2 + prev2[idx] + 1e-7f);
                prev2[idx] = o2;
            }
        }
    }
    if (doNH && useNH) {
        __shared__ float red[256];
        red[threadIdx.x] = r; __syncthreads();
        for (int s = 128; s > 0; s >>= 1) {
            if (threadIdx.x < (unsigned)s) red[threadIdx.x] += red[threadIdx.x + s];
            __syncthreads();
        }
        if (threadIdx.x == 0) partials[blockIdx.x] = red[0];
    }
}

__global__ __launch_bounds__(256)
void nh_reduce_kernel(const float* __restrict__ partials, int n, float* __restrict__ nh, float scale)
{
    float s = 0.f;
    for (int i = threadIdx.x; i < n; i += 256) s += partials[i];
    __shared__ float red[256];
    red[threadIdx.x] = s; __syncthreads();
    for (int st = 128; st > 0; st >>= 1) {
        if (threadIdx.x < (unsigned)st) red[threadIdx.x] += red[threadIdx.x + st];
        __syncthreads();
    }
    if (threadIdx.x == 0) nh[0] += red[0] * scale;
}

__global__ __launch_bounds__(256)
void zero_kernel(float* __restrict__ p, int n)
{
    int i = blockIdx.x * 256 + threadIdx.x;
    if (i < n) p[i] = 0.f;
}

__global__ __launch_bounds__(256)
void copy2_kernel(float* __restrict__ da, const float* __restrict__ sa,
                  float* __restrict__ db, const float* __restrict__ sb, int n)
{
    int i = blockIdx.x * 256 + threadIdx.x;
    if (i < n) { da[i] = sa[i]; db[i] = sb[i]; }
}

// final slice + ReLU + nhloss scalar
__global__ __launch_bounds__(256)
void finalize_kernel(const float* __restrict__ outs, int ldo, const float* __restrict__ nh,
                     float* __restrict__ out, int N, int NOUTF, int E)
{
    int idx = blockIdx.x * 256 + threadIdx.x;
    int total = N * NOUTF;
    if (idx < total) {
        int a = idx / NOUTF, c = idx - a * NOUTF;
        float v = outs[(size_t)a * ldo + 4 + c];
        if (c > 0) v = fmaxf(v, 0.f);
        out[idx] = v;
    }
    if (idx == 0) out[(size_t)N * NOUTF + E] = nh[0];
}

// ---------------------------------------------------------------------------
static inline int cdiv(int a, int b) { return (a + b - 1) / b; }

extern "C" void kernel_launch(void* const* d_in, const int* in_sizes, int n_in,
                              void* d_out, int out_size, void* d_ws, size_t ws_size,
                              hipStream_t stream)
{
    (void)in_sizes; (void)n_in; (void)out_size; (void)ws_size;
    const int N = 10000, E = 320000, F = 128, KR = 64;
    const int NOUT = 2005, NOUTF = 2001, NMOLc = 64;

    const float* R       = (const float*)d_in[0];
    const float* Mm      = (const float*)d_in[1];
    const float* QaAlpha = (const float*)d_in[2];
    const float* QaBeta  = (const float*)d_in[3];
    const float* em_W0   = (const float*)d_in[4];
    const float* em_b0   = (const float*)d_in[5];
    const float* em_W1   = (const float*)d_in[6];
    const float* em_b1   = (const float*)d_in[7];
    const float* em_W2   = (const float*)d_in[8];
    const float* em_b2   = (const float*)d_in[9];
    const float* iWrbf   = (const float*)d_in[10];
    const float* iWi     = (const float*)d_in[11];
    const float* ibi     = (const float*)d_in[12];
    const float* iWj     = (const float*)d_in[13];
    const float* ibj     = (const float*)d_in[14];
    const float* irW1    = (const float*)d_in[15];
    const float* irb1    = (const float*)d_in[16];
    const float* irW2    = (const float*)d_in[17];
    const float* irb2    = (const float*)d_in[18];
    const float* iWout   = (const float*)d_in[19];
    const float* ibout   = (const float*)d_in[20];
    const float* iu      = (const float*)d_in[21];
    const float* arW1    = (const float*)d_in[22];
    const float* arb1    = (const float*)d_in[23];
    const float* arW2    = (const float*)d_in[24];
    const float* arb2    = (const float*)d_in[25];
    const float* orW1    = (const float*)d_in[26];
    const float* orb1    = (const float*)d_in[27];
    const float* orW2    = (const float*)d_in[28];
    const float* orb2    = (const float*)d_in[29];
    const float* oW      = (const float*)d_in[30];
    const int*   Z       = (const int*)d_in[31];
    const int*   idx_i   = (const int*)d_in[32];
    const int*   idx_j   = (const int*)d_in[33];
    const int*   mol_idx = (const int*)d_in[34];

    float* ws = (float*)d_ws;
    size_t off = 0;
    float* rbf   = ws + off; off += (size_t)E * KR;
    float* g     = ws + off; off += (size_t)E * F;
    float* outs  = ws + off; off += (size_t)N * NOUT;
    float* tmpO  = ws + off; off += (size_t)N * NOUT;
    float* prev2 = ws + off; off += (size_t)N * NOUT;
    float* x     = ws + off; off += (size_t)N * F;
    float* h     = ws + off; off += (size_t)N * F;
    float* t1    = ws + off; off += (size_t)N * F;   // xi / m
    float* t2    = ws + off; off += (size_t)N * F;   // xj
    float* t3    = ws + off; off += (size_t)N * F;   // residual temp
    float* t4    = ws + off; off += (size_t)N * F;   // xo
    float* QaA   = ws + off; off += (size_t)N;
    float* QaB   = ws + off; off += (size_t)N;
    float* molAB = ws + off; off += 2 * NMOLc;       // QAlpha_mol, QBeta_mol
    float* molT  = ws + off; off += 4 * NMOLc;
    float* nh    = ws + off; off += 1;
    const int nblkO = cdiv(N * NOUT, 256);
    float* partials = ws + off; off += (size_t)nblkO;

    float* dout    = (float*)d_out;
    float* dij_out = dout + (size_t)N * NOUTF;

    auto gemm = [&](const float* X, const float* W, const float* bias,
                    const float* res, const float* iuv, float* Y,
                    int M, int Nn, int K, int ldw, int ldy, int flags) {
        dim3 gd(cdiv(M, 128), cdiv(Nn, 128));
        gemm_wmma_kernel<<<gd, 256, 0, stream>>>(X, W, bias, res, iuv, Y,
                                                 M, Nn, K, ldw, ldy, flags);
    };

    // --- preamble: edges/RBF, charge copies, molecule reference sums ---
    edge_rbf_kernel<<<cdiv(E, 256), 256, 0, stream>>>(R, idx_i, idx_j, rbf, dij_out, E);
    copy2_kernel<<<cdiv(N, 256), 256, 0, stream>>>(QaA, QaAlpha, QaB, QaBeta, N);
    segsum2_kernel<<<NMOLc, 256, 0, stream>>>(QaAlpha, QaBeta, molAB, N, NMOLc);
    zero_kernel<<<1, 256, 0, stream>>>(nh, 1);

    for (int j = 0; j <= 1; ++j) {
        // embedding: h=ssp(f@W0+b0); h=ssp(h@W1+b1); x=h@W2+b2
        embed_kernel<<<cdiv(N * F, 256), 256, 0, stream>>>(Z, Mm, QaA, QaB, em_W0, em_b0, h, N);
        gemm(h,  em_W1, em_b1, nullptr, nullptr, t1, N, F, F, F, F, F_SSP_OUT);
        gemm(t1, em_W2, em_b2, nullptr, nullptr, x,  N, F, F, F, F, 0);
        zero_kernel<<<nblkO, 256, 0, stream>>>(outs, N * NOUT);

        for (int i = 0; i < 3; ++i) {
            // g = rbf @ iWrbf[i]   (K=64)
            gemm(rbf, iWrbf + (size_t)i * KR * F, nullptr, nullptr, nullptr,
                 g, E, F, KR, F, F, 0);
            // xi, xj
            gemm(x, iWi + (size_t)i * F * F, ibi + i * F, nullptr, nullptr,
                 t1, N, F, F, F, F, F_SSP_IN | F_SSP_OUT);
            gemm(x, iWj + (size_t)i * F * F, ibj + i * F, nullptr, nullptr,
                 t2, N, F, F, F, F, F_SSP_IN | F_SSP_OUT);
            // m = xi + segment_sum(g * xj[idx_j])
            message_kernel<<<N, 128, 0, stream>>>(t1, g, t2, idx_j, N);
            // interaction residuals
            for (int r = 0; r < 2; ++r) {
                const float* W1 = irW1 + (size_t)(i * 2 + r) * F * F;
                const float* b1 = irb1 + (size_t)(i * 2 + r) * F;
                const float* W2 = irW2 + (size_t)(i * 2 + r) * F * F;
                const float* b2 = irb2 + (size_t)(i * 2 + r) * F;
                gemm(t1, W1, b1, nullptr, nullptr, t3, N, F, F, F, F, F_SSP_IN);
                gemm(t3, W2, b2, t1, nullptr, t1, N, F, F, F, F, F_SSP_IN | F_RES);
            }
            // x = iu*x + ssp(m) @ iWout + ibout
            gemm(t1, iWout + (size_t)i * F * F, ibout + i * F, x, iu + i * F,
                 x, N, F, F, F, F, F_SSP_IN | F_RES | F_RES_IU);
            // atom residuals
            for (int r = 0; r < 2; ++r) {
                const float* W1 = arW1 + (size_t)(i * 2 + r) * F * F;
                const float* b1 = arb1 + (size_t)(i * 2 + r) * F;
                const float* W2 = arW2 + (size_t)(i * 2 + r) * F * F;
                const float* b2 = arb2 + (size_t)(i * 2 + r) * F;
                gemm(x,  W1, b1, nullptr, nullptr, t3, N, F, F, F, F, F_SSP_IN);
                gemm(t3, W2, b2, x, nullptr, x, N, F, F, F, F, F_SSP_IN | F_RES);
            }
            // output residual: xo = x + ssp(ssp(x)@orW1+orb1)@orW2+orb2
            gemm(x,  orW1 + (size_t)i * F * F, orb1 + i * F, nullptr, nullptr,
                 t3, N, F, F, F, F, F_SSP_IN);
            gemm(t3, orW2 + (size_t)i * F * F, orb2 + i * F, x, nullptr,
                 t4, N, F, F, F, F, F_SSP_IN | F_RES);
            // out = ssp(xo) @ oW[i]  (wide N=2005)
            gemm(t4, oW + (size_t)i * F * NOUT, nullptr, nullptr, nullptr,
                 tmpO, N, NOUT, F, NOUT, NOUT, F_SSP_IN);
            // accumulate + nh-loss bookkeeping
            int doNH  = (j == 1) ? 1 : 0;
            int useNH = (j == 1 && i > 0) ? 1 : 0;
            accum_out_kernel<<<nblkO, 256, 0, stream>>>(outs, tmpO, prev2, partials,
                                                        N * NOUT, NOUT, doNH, useNH);
            if (useNH)
                nh_reduce_kernel<<<1, 256, 0, stream>>>(partials, nblkO, nh,
                                                        1.f / ((float)(N - 4) * (float)NOUT));
        }
        // SCC charge update
        segsum_cols_kernel<<<NMOLc, 256, 0, stream>>>(outs, NOUT, molT, N, NMOLc);
        scc_update_kernel<<<cdiv(N, 256), 256, 0, stream>>>(QaA, QaB, outs, NOUT,
                                                            molAB, molT, mol_idx, N, NMOLc);
    }

    finalize_kernel<<<cdiv(N * NOUTF, 256), 256, 0, stream>>>(outs, NOUT, nh, dout,
                                                              N, NOUTF, E);
}
